// Encoder_90314572300663
// MI455X (gfx1250) — matrix operand order repaired and checked by reading, not compiled, not run on hardware
//
#include <hip/hip_runtime.h>
#include <hip/hip_bf16.h>
#include <math.h>

// ---------------------------------------------------------------------------
// Problem constants (from reference): L=6, B=16, S=512, D=512, SD=128, E=1536
// ---------------------------------------------------------------------------
#define LAYERS 6
#define BATCH  16
#define SEQ    512
#define DMODEL 512
#define SDIM   128
#define EDIM   1536
#define MROWS  (BATCH * SEQ)      // 8192 token rows

typedef __bf16 v16bf __attribute__((ext_vector_type(16)));
typedef float  v8f   __attribute__((ext_vector_type(8)));

union FragAB { uint4 u[2]; v16bf v; };

// CDNA5 async global->LDS path (ASYNCcnt), with synchronous fallback.
#if defined(__has_builtin)
#if __has_builtin(__builtin_amdgcn_global_load_async_to_lds_b128) && \
    __has_builtin(__builtin_amdgcn_s_wait_asynccnt)
#define USE_ASYNC 1
#endif
#endif
#ifndef USE_ASYNC
#define USE_ASYNC 0
#endif

#if USE_ASYNC
// Builtin signature (from hipcc diagnostics): first arg is a pointer to a
// 16-byte int vector in the global (__device__/AS1) address space; LDS side
// is the AS3 equivalent.
typedef int v4i __attribute__((vector_size(16)));
typedef __attribute__((address_space(1))) v4i* gaddr4_t;
typedef __attribute__((address_space(3))) v4i* laddr4_t;
#endif

enum {
    MODE_BF16  = 0,   // outH[m*N+n]        = bf16(acc + bias[n])          (u)
    MODE_BF16T = 1,   // outH[b*sO + n*ldT + t] = bf16(acc + bias[n])      (v^T)
    MODE_F32   = 2,   // outF[m*N+n]        = acc + bias[n]                (z)
    MODE_QK    = 3,   // relu(acc)^2, masked, -> bf16 (batched)            (a)
    MODE_AVU   = 4,   // outH[idx] = bf16(acc * uH[idx]) (in-place over u) (w)
    MODE_RESID = 5    // outF[m*N+n] += acc + bias[n]                      (x)
};

// ---------------------------------------------------------------------------
// Generic bf16 WMMA GEMM:  C(M x N) = A(M x K, row-major bf16) * Bt^T
// Bt is stored N-major (N x K row-major) so both operands stage identically.
// Block = 256 threads (8 waves), tile 128(M) x 64(N), K stepped by 64 through
// a double-buffered LDS stage filled with async global->LDS copies.
// Each wave: 2x2 quad of 16x16 fragments, 8 WMMAs per K-stage.
// Requires: M_per_batch % 128 == 0, N % 64 == 0, K % 64 == 0 (all hold here).
// ---------------------------------------------------------------------------
__global__ __launch_bounds__(256) void gemm_bf16_kernel(
    const __hip_bfloat16* __restrict__ A,  long long strideA, int lda,
    const __hip_bfloat16* __restrict__ Bt, long long strideB, int ldb,
    int K, int N,
    const float* __restrict__ bias,
    float* __restrict__ outF,
    __hip_bfloat16* __restrict__ outH,
    long long strideO,
    const __hip_bfloat16* __restrict__ uH,
    const unsigned char* __restrict__ mask,
    int mode, int rpb, int ldT)
{
    __shared__ uint4 smA4[2][1024];   // 2 x (128 rows x 64 bf16) = 32 KB
    __shared__ uint4 smB4[2][512];    // 2 x ( 64 rows x 64 bf16) = 16 KB

    const int t  = threadIdx.x;
    const int zb = blockIdx.z;
    const __hip_bfloat16* Ab = A  + (long long)zb * strideA
                                  + (long long)blockIdx.y * 128 * lda;
    const __hip_bfloat16* Bb = Bt + (long long)zb * strideB
                                  + (long long)blockIdx.x * 64 * ldb;

    const int lane = t & 31;
    const int wave = t >> 5;
    const int wm   = (wave & 3) * 32;   // wave M offset in tile (0..96)
    const int wn   = (wave >> 2) * 32;  // wave N offset in tile (0 or 32)
    const int half = lane >> 4;
    const int sub  = lane & 15;

    v8f c[2][2] = {};

    // Stage one K-chunk (64) of A (128x64) and B (64x64) into LDS buffer `buf`.
    // 6 x 16B per thread: 4 for A, 2 for B.
    auto stage = [&](int buf, int kc) {
#if USE_ASYNC
        #pragma unroll
        for (int i = 0; i < 4; ++i) {
            int idx = t + i * 256, row = idx >> 3, seg = idx & 7;
            const __hip_bfloat16* g = Ab + (long long)row * lda + kc + seg * 8;
            __builtin_amdgcn_global_load_async_to_lds_b128(
                (gaddr4_t)g, (laddr4_t)&smA4[buf][idx], 0, 0);
        }
        #pragma unroll
        for (int i = 0; i < 2; ++i) {
            int idx = t + i * 256, row = idx >> 3, seg = idx & 7;
            const __hip_bfloat16* g = Bb + (long long)row * ldb + kc + seg * 8;
            __builtin_amdgcn_global_load_async_to_lds_b128(
                (gaddr4_t)g, (laddr4_t)&smB4[buf][idx], 0, 0);
        }
#else
        #pragma unroll
        for (int i = 0; i < 4; ++i) {
            int idx = t + i * 256, row = idx >> 3, seg = idx & 7;
            smA4[buf][idx] = ((const uint4*)(Ab + (long long)row * lda + kc))[seg];
        }
        #pragma unroll
        for (int i = 0; i < 2; ++i) {
            int idx = t + i * 256, row = idx >> 3, seg = idx & 7;
            smB4[buf][idx] = ((const uint4*)(Bb + (long long)row * ldb + kc))[seg];
        }
        if (kc + 64 < K) {
            __builtin_prefetch(Ab + (long long)(t >> 1) * lda + kc + 64, 0, 1);
        }
#endif
    };

    const int nIter = K >> 6;   // K / 64
    stage(0, 0);

    for (int it = 0; it < nIter; ++it) {
        const int cur  = it & 1;
        const bool more = (it + 1 < nIter);
        if (more) stage(cur ^ 1, (it + 1) << 6);
#if USE_ASYNC
        if (more) __builtin_amdgcn_s_wait_asynccnt(6);  // allow just-issued batch
        else      __builtin_amdgcn_s_wait_asynccnt(0);
#endif
        __syncthreads();

        #pragma unroll
        for (int kh = 0; kh < 2; ++kh) {
            // A frag layout (ISA 16-bit A 16x32): lanes 0-15 hold K{0..7,16..23},
            // lanes 16-31 K{8..15,24..31}; B frag: same K order as A,
            // column = sub.
            FragAB a0, a1, b0, b1;
            { const uint4* p = &smA4[cur][(wm +  0 + sub) * 8];
              a0.u[0] = p[kh * 4 + half];     a0.u[1] = p[kh * 4 + 2 + half]; }
            { const uint4* p = &smA4[cur][(wm + 16 + sub) * 8];
              a1.u[0] = p[kh * 4 + half];     a1.u[1] = p[kh * 4 + 2 + half]; }
            { const uint4* p = &smB4[cur][(wn +  0 + sub) * 8];
              b0.u[0] = p[kh * 4 + half];     b0.u[1] = p[kh * 4 + 2 + half]; }
            { const uint4* p = &smB4[cur][(wn + 16 + sub) * 8];
              b1.u[0] = p[kh * 4 + half];     b1.u[1] = p[kh * 4 + 2 + half]; }

            c[0][0] = __builtin_amdgcn_wmma_f32_16x16x32_bf16(false, a0.v, false, b0.v, (short)0, c[0][0], false, false);
            c[0][1] = __builtin_amdgcn_wmma_f32_16x16x32_bf16(false, a0.v, false, b1.v, (short)0, c[0][1], false, false);
            c[1][0] = __builtin_amdgcn_wmma_f32_16x16x32_bf16(false, a1.v, false, b0.v, (short)0, c[1][0], false, false);
            c[1][1] = __builtin_amdgcn_wmma_f32_16x16x32_bf16(false, a1.v, false, b1.v, (short)0, c[1][1], false, false);
        }
        __syncthreads();   // protect buffer `cur` before it is re-staged
    }

    // ---- epilogue: C frag layout = VGPR r -> M = r + 8*half, N = sub ----
    #pragma unroll
    for (int fi = 0; fi < 2; ++fi)
    #pragma unroll
    for (int fj = 0; fj < 2; ++fj)
    #pragma unroll
    for (int r = 0; r < 8; ++r) {
        long long m = (long long)blockIdx.y * 128 + wm + fi * 16 + half * 8 + r;
        long long n = (long long)blockIdx.x * 64  + wn + fj * 16 + sub;
        float acc = c[fi][fj][r];
        if (mode == MODE_BF16) {
            outH[m * N + n] = __float2bfloat16(acc + bias[n]);
        } else if (mode == MODE_BF16T) {
            long long bb = m / rpb, tt = m % rpb;
            outH[bb * strideO + n * ldT + tt] = __float2bfloat16(acc + bias[n]);
        } else if (mode == MODE_F32) {
            outF[m * N + n] = acc + bias[n];
        } else if (mode == MODE_QK) {
            float rl  = acc > 0.f ? acc : 0.f;
            float val = rl * rl;
            if (mask[(long long)zb * N + n]) val = 0.f;
            outH[(long long)zb * strideO + m * N + n] = __float2bfloat16(val);
        } else if (mode == MODE_AVU) {
            long long gm  = (long long)zb * rpb + m;
            long long idx = gm * N + n;
            outH[idx] = __float2bfloat16(acc * __bfloat162float(uH[idx]));
        } else { // MODE_RESID
            outF[m * N + n] += acc + bias[n];
        }
    }
}

// ---------------------------------------------------------------------------
// x[b,s,:] = embed[tok[b,s]] * sqrt(D) + sinusoidal PE(s, :)
// ---------------------------------------------------------------------------
__global__ __launch_bounds__(256) void embed_kernel(
    const int* __restrict__ tok, const float* __restrict__ embed,
    float* __restrict__ x)
{
    long long i = (long long)blockIdx.x * 256 + threadIdx.x;  // < MROWS*DMODEL
    int m = (int)(i >> 9);
    int d = (int)(i & 511);
    int s = m & (SEQ - 1);
    float val = embed[(long long)tok[m] * DMODEL + d] * 22.62741699796952f;
    float e   = -2.0f * (float)(d >> 1) * (1.0f / (float)DMODEL);
    float ang = (float)s * __powf(10000.0f, e);
    val += ((d & 1) == 0) ? __sinf(ang) : __cosf(ang);
    x[i] = val;
}

// ---------------------------------------------------------------------------
// Transpose + cast: in (K x N) f32 row-major  ->  out (N x K) bf16 row-major
// ---------------------------------------------------------------------------
__global__ __launch_bounds__(256) void convertT_kernel(
    const float* __restrict__ in, __hip_bfloat16* __restrict__ out,
    int K, int N)
{
    long long i = (long long)blockIdx.x * 256 + threadIdx.x;
    if (i < (long long)K * N) {
        int k = (int)(i / N), n = (int)(i % N);
        out[(long long)n * K + k] = __float2bfloat16(in[i]);
    }
}

// ---------------------------------------------------------------------------
// LayerNorm over D=512, one 256-thread block per row -> bf16 xn
// ---------------------------------------------------------------------------
__global__ __launch_bounds__(256) void ln_kernel(
    const float* __restrict__ x, const float* __restrict__ g,
    const float* __restrict__ b, __hip_bfloat16* __restrict__ out)
{
    __shared__ float s1[256], s2[256];
    const int row = blockIdx.x, t = threadIdx.x;
    const float* xr = x + (long long)row * DMODEL;
    float v0 = xr[t], v1 = xr[t + 256];
    s1[t] = v0 + v1;
    s2[t] = v0 * v0 + v1 * v1;
    __syncthreads();
    for (int off = 128; off > 0; off >>= 1) {
        if (t < off) { s1[t] += s1[t + off]; s2[t] += s2[t + off]; }
        __syncthreads();
    }
    float mean = s1[0] * (1.0f / DMODEL);
    float var  = s2[0] * (1.0f / DMODEL) - mean * mean;
    float inv  = rsqrtf(var + 1e-5f);
    __hip_bfloat16* o = out + (long long)row * DMODEL;
    o[t]       = __float2bfloat16((v0 - mean) * inv * g[t]       + b[t]);
    o[t + 256] = __float2bfloat16((v1 - mean) * inv * g[t + 256] + b[t + 256]);
}

// Final LN -> f32 output
__global__ __launch_bounds__(256) void ln_final_kernel(
    const float* __restrict__ x, const float* __restrict__ g,
    const float* __restrict__ b, float* __restrict__ out)
{
    __shared__ float s1[256], s2[256];
    const int row = blockIdx.x, t = threadIdx.x;
    const float* xr = x + (long long)row * DMODEL;
    float v0 = xr[t], v1 = xr[t + 256];
    s1[t] = v0 + v1;
    s2[t] = v0 * v0 + v1 * v1;
    __syncthreads();
    for (int off = 128; off > 0; off >>= 1) {
        if (t < off) { s1[t] += s1[t + off]; s2[t] += s2[t + off]; }
        __syncthreads();
    }
    float mean = s1[0] * (1.0f / DMODEL);
    float var  = s2[0] * (1.0f / DMODEL) - mean * mean;
    float inv  = rsqrtf(var + 1e-5f);
    float* o = out + (long long)row * DMODEL;
    o[t]       = (v0 - mean) * inv * g[t]       + b[t];
    o[t + 256] = (v1 - mean) * inv * g[t + 256] + b[t + 256];
}

// ---------------------------------------------------------------------------
// q = z*gq + bq, k = z*gk + bk  (g/b indexed by in-sequence position)
// ---------------------------------------------------------------------------
__global__ __launch_bounds__(256) void qkprep_kernel(
    const float* __restrict__ z,
    const float* __restrict__ gq, const float* __restrict__ bq,
    const float* __restrict__ gk, const float* __restrict__ bk,
    __hip_bfloat16* __restrict__ q, __hip_bfloat16* __restrict__ k)
{
    long long i = (long long)blockIdx.x * 256 + threadIdx.x;  // < MROWS*SDIM
    int m  = (int)(i >> 7);
    int sd = (int)(i & 127);
    int s  = m & (SEQ - 1);
    long long gi = (long long)s * SDIM + sd;
    float zv = z[i];
    q[i] = __float2bfloat16(zv * gq[gi] + bq[gi]);
    k[i] = __float2bfloat16(zv * gk[gi] + bk[gi]);
}

// ---------------------------------------------------------------------------
// Host launcher
// ---------------------------------------------------------------------------
extern "C" void kernel_launch(void* const* d_in, const int* in_sizes, int n_in,
                              void* d_out, int out_size, void* d_ws, size_t ws_size,
                              hipStream_t stream)
{
    (void)in_sizes; (void)n_in; (void)out_size; (void)ws_size;

    const int*           tok   = (const int*)d_in[0];
    const unsigned char* mask  = (const unsigned char*)d_in[1];  // bool
    const float* embed = (const float*)d_in[2];
    const float* ln_g  = (const float*)d_in[3];
    const float* ln_b  = (const float*)d_in[4];
    const float* Wz    = (const float*)d_in[5];
    const float* bz    = (const float*)d_in[6];
    const float* gq    = (const float*)d_in[7];
    const float* bq    = (const float*)d_in[8];
    const float* gk    = (const float*)d_in[9];
    const float* bk    = (const float*)d_in[10];
    const float* Wu    = (const float*)d_in[11];
    const float* bu    = (const float*)d_in[12];
    const float* Wv    = (const float*)d_in[13];
    const float* bv    = (const float*)d_in[14];
    const float* Wo    = (const float*)d_in[15];
    const float* bo    = (const float*)d_in[16];
    const float* fin_g = (const float*)d_in[17];
    const float* fin_b = (const float*)d_in[18];

    // ---- carve workspace ----
    size_t off = 0;
    auto alloc = [&](size_t bytes) {
        void* p = (char*)d_ws + off;
        off += (bytes + 255) & ~(size_t)255;
        return p;
    };
    float*          x   = (float*)alloc((size_t)MROWS * DMODEL * 4);          // 16 MB
    __hip_bfloat16* xn  = (__hip_bfloat16*)alloc((size_t)MROWS * DMODEL * 2); //  8 MB
    __hip_bfloat16* u   = (__hip_bfloat16*)alloc((size_t)MROWS * EDIM * 2);   // 24 MB (reused for w)
    __hip_bfloat16* vT  = (__hip_bfloat16*)alloc((size_t)BATCH * EDIM * SEQ * 2); // 24 MB
    float*          z   = (float*)alloc((size_t)MROWS * SDIM * 4);            //  4 MB
    __hip_bfloat16* qb  = (__hip_bfloat16*)alloc((size_t)MROWS * SDIM * 2);   //  2 MB
    __hip_bfloat16* kb  = (__hip_bfloat16*)alloc((size_t)MROWS * SDIM * 2);   //  2 MB
    __hip_bfloat16* ab  = (__hip_bfloat16*)alloc((size_t)BATCH * SEQ * SEQ * 2); // 8 MB
    __hip_bfloat16* WuT = (__hip_bfloat16*)alloc((size_t)EDIM * DMODEL * 2);  // 1.5 MB
    __hip_bfloat16* WvT = (__hip_bfloat16*)alloc((size_t)EDIM * DMODEL * 2);
    __hip_bfloat16* WoT = (__hip_bfloat16*)alloc((size_t)DMODEL * EDIM * 2);
    __hip_bfloat16* WzT = (__hip_bfloat16*)alloc((size_t)SDIM * DMODEL * 2);

    // ---- embedding + positional encoding ----
    embed_kernel<<<(MROWS * DMODEL) / 256, 256, 0, stream>>>(tok, embed, x);

    for (int l = 0; l < LAYERS; ++l) {
        const float* Wz_l = Wz + (long long)l * DMODEL * SDIM;
        const float* Wu_l = Wu + (long long)l * DMODEL * EDIM;
        const float* Wv_l = Wv + (long long)l * DMODEL * EDIM;
        const float* Wo_l = Wo + (long long)l * EDIM * DMODEL;
        const float* bz_l = bz + (long long)l * SDIM;
        const float* bu_l = bu + (long long)l * EDIM;
        const float* bv_l = bv + (long long)l * EDIM;
        const float* bo_l = bo + (long long)l * DMODEL;
        const float* gq_l = gq + (long long)l * SEQ * SDIM;
        const float* bq_l = bq + (long long)l * SEQ * SDIM;
        const float* gk_l = gk + (long long)l * SEQ * SDIM;
        const float* bk_l = bk + (long long)l * SEQ * SDIM;

        // weights -> bf16, transposed to N-major
        convertT_kernel<<<(DMODEL * EDIM + 255) / 256, 256, 0, stream>>>(Wu_l, WuT, DMODEL, EDIM);
        convertT_kernel<<<(DMODEL * EDIM + 255) / 256, 256, 0, stream>>>(Wv_l, WvT, DMODEL, EDIM);
        convertT_kernel<<<(EDIM * DMODEL + 255) / 256, 256, 0, stream>>>(Wo_l, WoT, EDIM, DMODEL);
        convertT_kernel<<<(DMODEL * SDIM + 255) / 256, 256, 0, stream>>>(Wz_l, WzT, DMODEL, SDIM);

        // xn = LN(x)
        ln_kernel<<<MROWS, 256, 0, stream>>>(x, ln_g + (long long)l * DMODEL,
                                             ln_b + (long long)l * DMODEL, xn);

        // u = xn @ Wu + bu  (bf16)
        gemm_bf16_kernel<<<dim3(EDIM / 64, MROWS / 128, 1), 256, 0, stream>>>(
            xn, 0, DMODEL, WuT, 0, DMODEL, DMODEL, EDIM,
            bu_l, nullptr, u, 0, nullptr, nullptr, MODE_BF16, 0, 0);

        // v^T = (xn @ Wv + bv)^T per batch  (bf16, N-major for a@v)
        gemm_bf16_kernel<<<dim3(EDIM / 64, MROWS / 128, 1), 256, 0, stream>>>(
            xn, 0, DMODEL, WvT, 0, DMODEL, DMODEL, EDIM,
            bv_l, nullptr, vT, (long long)EDIM * SEQ, nullptr, nullptr,
            MODE_BF16T, SEQ, SEQ);

        // z = xn @ Wz + bz  (f32)
        gemm_bf16_kernel<<<dim3(SDIM / 64, MROWS / 128, 1), 256, 0, stream>>>(
            xn, 0, DMODEL, WzT, 0, DMODEL, DMODEL, SDIM,
            bz_l, z, nullptr, 0, nullptr, nullptr, MODE_F32, 0, 0);

        // q,k = z*g + b  (bf16)
        qkprep_kernel<<<(MROWS * SDIM) / 256, 256, 0, stream>>>(
            z, gq_l, bq_l, gk_l, bk_l, qb, kb);

        // a = mask(relu(q @ k^T)^2)  (batched over B)
        gemm_bf16_kernel<<<dim3(SEQ / 64, SEQ / 128, BATCH), 256, 0, stream>>>(
            qb, (long long)SEQ * SDIM, SDIM, kb, (long long)SEQ * SDIM, SDIM,
            SDIM, SEQ, nullptr, nullptr, ab, (long long)SEQ * SEQ,
            nullptr, mask, MODE_QK, 0, 0);

        // w = u * (a @ v)  (batched; written in place over u)
        gemm_bf16_kernel<<<dim3(EDIM / 64, SEQ / 128, BATCH), 256, 0, stream>>>(
            ab, (long long)SEQ * SEQ, SEQ, vT, (long long)EDIM * SEQ, SEQ,
            SEQ, EDIM, nullptr, nullptr, u, 0, u, nullptr, MODE_AVU, SEQ, 0);

        // x += w @ Wo + bo
        gemm_bf16_kernel<<<dim3(DMODEL / 64, MROWS / 128, 1), 256, 0, stream>>>(
            u, 0, EDIM, WoT, 0, EDIM, EDIM, DMODEL,
            bo_l, x, nullptr, 0, nullptr, nullptr, MODE_RESID, 0, 0);
    }

    // final LN -> d_out (f32)
    ln_final_kernel<<<MROWS, 256, 0, stream>>>(x, fin_g, fin_b, (float*)d_out);
}
